// NeradMB_35983236006532
// MI455X (gfx1250) — compile-verified, matrix operand order checked by reading, not run on hardware
//
#include <hip/hip_runtime.h>
#include <stdint.h>

#define DD 6
#define RRD 5
#define HDIM 256
#define THREADS 256
#define WPB (THREADS / 32)      // waves per block
#define RPW 16                  // rays per wave

typedef _Float16 v8h  __attribute__((ext_vector_type(8)));
typedef _Float16 v16h __attribute__((ext_vector_type(16)));
typedef float    v8f  __attribute__((ext_vector_type(8)));

union V16U { v16h v; v8h h[2]; };

// global-address-space pointer types (keep global_load codegen after laundering)
typedef const __attribute__((address_space(1))) _Float16* ghp;
typedef const __attribute__((address_space(1))) float*    gfp;
typedef const __attribute__((address_space(1))) v8h*      gv8hp;
typedef const __attribute__((address_space(1))) v8f*      gv8fp;

// ---------------- weight prep: transpose + f16 convert + pad ----------------
__global__ void prep_w0_k(const float* __restrict__ W0, _Float16* __restrict__ W0t) {
  int idx = blockIdx.x * blockDim.x + threadIdx.x;       // 256 x 32 (K padded 12->32)
  if (idx >= 256 * 32) return;
  int n = idx >> 5, k = idx & 31;
  float v = (k < 12) ? W0[k * 256 + n] : 0.0f;
  W0t[n * 32 + k] = (_Float16)v;
}
__global__ void prep_w_k(const float* __restrict__ W, _Float16* __restrict__ Wt) {
  int idx = blockIdx.x * blockDim.x + threadIdx.x;       // 256 x 256
  if (idx >= 256 * 256) return;
  int n = idx >> 8, k = idx & 255;
  Wt[n * 256 + k] = (_Float16)W[k * 256 + n];
}
__global__ void prep_w3_k(const float* __restrict__ W3, _Float16* __restrict__ W3t) {
  int idx = blockIdx.x * blockDim.x + threadIdx.x;       // 16 x 256 (Nout padded 3->16)
  if (idx >= 16 * 256) return;
  int n = idx >> 8, k = idx & 255;
  float v = (n < 3) ? W3[k * 3 + n] : 0.0f;
  W3t[n * 256 + k] = (_Float16)v;
}

// ---- pointer laundering: defeats LICM hoisting of loop-invariant weight
//      loads out of the bounce loop (prevents multi-KB scratch spills), while
//      the address_space(1) result keeps true global_load codegen (no flat).
static __device__ __forceinline__ ghp lh(const _Float16* p) {
  unsigned long long x = (unsigned long long)p;
  asm("" : "+s"(x));
  return (ghp)x;
}
static __device__ __forceinline__ gfp lf(const float* p) {
  unsigned long long x = (unsigned long long)p;
  asm("" : "+s"(x));
  return (gfp)x;
}

// relu (folds to max-with-zero)
static __device__ __forceinline__ float relu(float x) {
  return __builtin_amdgcn_fmed3f(x, 0.0f, __builtin_inff());
}

// A-operand (weight) fragment: row = out-neuron, two contiguous 16B K-chunks
static __device__ __forceinline__ v16h load_wfrag(ghp wrow, int koff) {
  V16U a;
  a.h[0] = *(gv8hp)(wrow + koff);
  a.h[1] = *(gv8hp)(wrow + koff + 16);
  return a.v;
}

#define WMMA(A, B, C) __builtin_amdgcn_wmma_f32_16x16x32_f16(false, (A), false, (B), (short)0, (C), false, false)

// hidden layer: Act(256 x 16 rays) -> Act(256 x 16 rays), fully in registers.
// D-fragment of tile pair (2s,2s+1) packs directly into Bout[s]
// (identical lane/VGPR layout), relu + f16 convert in-register.
static __device__ __forceinline__ void layer256(ghp Wt, gfp bias,
                                                const v16h* __restrict__ Bin,
                                                v16h* __restrict__ Bout,
                                                int rowSel, int halfSel)
{
  #pragma unroll
  for (int so = 0; so < 8; ++so) {
    v8f accA = *(gv8fp)(bias + (2 * so) * 16 + halfSel);
    v8f accB = *(gv8fp)(bias + (2 * so + 1) * 16 + halfSel);
    ghp wrowA = Wt + ((2 * so) * 16 + rowSel) * HDIM;
    ghp wrowB = Wt + ((2 * so + 1) * 16 + rowSel) * HDIM;
    #pragma unroll
    for (int s = 0; s < 8; ++s) {
      v16h a0 = load_wfrag(wrowA, s * 32 + halfSel);
      v16h a1 = load_wfrag(wrowB, s * 32 + halfSel);
      accA = WMMA(a0, Bin[s], accA);
      accB = WMMA(a1, Bin[s], accB);
    }
    V16U nb;
    #pragma unroll
    for (int j = 0; j < 8; ++j) {
      nb.v[j]     = (_Float16)relu(accA[j]);
      nb.v[8 + j] = (_Float16)relu(accB[j]);
    }
    Bout[so] = nb.v;
  }
}

// input layer: features (K=32 padded) -> 256 activations
static __device__ __forceinline__ void layer0(ghp W0t, gfp b0,
                                              v16h Bfeat, v16h* __restrict__ Bout,
                                              int rowSel, int halfSel)
{
  #pragma unroll
  for (int so = 0; so < 8; ++so) {
    v8f accA = *(gv8fp)(b0 + (2 * so) * 16 + halfSel);
    v8f accB = *(gv8fp)(b0 + (2 * so + 1) * 16 + halfSel);
    v16h a0 = load_wfrag(W0t + ((2 * so) * 16 + rowSel) * 32, halfSel);
    v16h a1 = load_wfrag(W0t + ((2 * so + 1) * 16 + rowSel) * 32, halfSel);
    accA = WMMA(a0, Bfeat, accA);
    accB = WMMA(a1, Bfeat, accB);
    V16U nb;
    #pragma unroll
    for (int j = 0; j < 8; ++j) {
      nb.v[j]     = (_Float16)relu(accA[j]);
      nb.v[8 + j] = (_Float16)relu(accB[j]);
    }
    Bout[so] = nb.v;
  }
}

// output layer: 256 -> 16 (3 real); lanes 0-15 end with rgb in acc[0..2]
static __device__ __forceinline__ v8f layer3(ghp W3t, gfp b3,
                                             const v16h* __restrict__ Bin,
                                             int rowSel, int halfSel)
{
  v8f acc;
  #pragma unroll
  for (int r = 0; r < 8; ++r) acc[r] = 0.f;
  acc[0] = (halfSel == 0) ? b3[0] : 0.f;
  acc[1] = (halfSel == 0) ? b3[1] : 0.f;
  acc[2] = (halfSel == 0) ? b3[2] : 0.f;
  ghp wrow = W3t + rowSel * HDIM;
  #pragma unroll
  for (int s = 0; s < 8; ++s) {
    v16h a = load_wfrag(wrow, s * 32 + halfSel);
    acc = WMMA(a, Bin[s], acc);
  }
  return acc;
}

// -------------------- fused 6-bounce, register-resident ---------------------
__global__ __launch_bounds__(THREADS)
void nerad_reg_k(const float* __restrict__ si_p, const float* __restrict__ si_wo,
                 const float* __restrict__ si_n, const float* __restrict__ albedo,
                 const float* __restrict__ em_hit, const float* __restrict__ em_sample,
                 const float* __restrict__ bsdf_weight, const float* __restrict__ eta_s,
                 const float* __restrict__ rr_u,
                 const float* __restrict__ b0, const float* __restrict__ b1,
                 const float* __restrict__ b2, const float* __restrict__ b3,
                 const _Float16* __restrict__ W0t, const _Float16* __restrict__ W1t,
                 const _Float16* __restrict__ W2t, const _Float16* __restrict__ W3t,
                 const unsigned char* __restrict__ si_valid,
                 const unsigned char* __restrict__ null_flag,
                 float* __restrict__ out, int N)
{
  const int lane    = threadIdx.x & 31;
  const int wave    = threadIdx.x >> 5;
  const int rowSel  = lane & 15;
  const int halfSel = (lane >> 4) << 3;                  // 0 or 8
  const int rayBase = (blockIdx.x * WPB + wave) * RPW;
  const int ray     = rayBase + rowSel;                  // each half-wave mirrors 16 rays
  const bool owner  = (lane < 16);                       // lanes holding per-ray state

  if (threadIdx.x == 0) { __builtin_prefetch(W1t, 0, 3); __builtin_prefetch(W2t, 0, 3); }

  v16h Bcur[8], Bnxt[8];

  // per-ray path state (meaningful on owner lanes)
  float res0 = 0.f, res1 = 0.f, res2 = 0.f;
  float thr0 = 1.f, thr1 = 1.f, thr2 = 1.f;
  float L0 = 0.f, L1 = 0.f, L2 = 0.f;
  float eta = 1.f;
  int depth = 0;
  bool active = true, validray = false;

  #pragma unroll 1
  for (int d = 0; d < DD; ++d) {
    // fresh (laundered) global pointers each bounce: weight loads re-issued
    // from L2-resident global memory, never hoisted+spilled to scratch.
    ghp w0t = lh(W0t);
    ghp w1t = lh(W1t);
    ghp w2t = lh(W2t);
    ghp w3t = lh(W3t);
    gfp bb0 = lf(b0);
    gfp bb1 = lf(b1);
    gfp bb2 = lf(b2);
    gfp bb3 = lf(b3);

    // ---- feature B-fragment: lane holds features halfSel..halfSel+7 of its
    //      ray; K 16..31 is zero padding (matches zero-padded W0t rows) ----
    V16U f;
    #pragma unroll
    for (int j = 0; j < 8; ++j) {
      const int idx = halfSel + j;
      float v = 0.f;
      if (idx < 12 && ray < N) {
        const float* src = (idx < 3) ? si_p : (idx < 6) ? si_wo : (idx < 9) ? si_n : albedo;
        const int comp = idx - ((idx < 3) ? 0 : (idx < 6) ? 3 : (idx < 9) ? 6 : 9);
        v = src[((long long)d * N + ray) * 3 + comp];
      }
      f.v[j]     = (_Float16)v;
      f.v[8 + j] = (_Float16)0.f;
    }

    // ---- MLP, all activations in registers ----
    layer0  (w0t, bb0, f.v, Bcur, rowSel, halfSel);
    layer256(w1t, bb1, Bcur, Bnxt, rowSel, halfSel);
    layer256(w2t, bb2, Bnxt, Bcur, rowSel, halfSel);
    v8f c = layer3(w3t, bb3, Bcur, rowSel, halfSel);

    // ---- per-ray path-tracing state update (owner lanes only) ----
    if (owner && ray < N) {
      const long long gi = (long long)d * N + ray;
      const bool valid = si_valid[gi] != 0;
      const bool nullf = null_flag[gi] != 0;
      float c0 = c[0], c1 = c[1], c2 = c[2];
      if (!(active && valid)) { c0 = 0.f; c1 = 0.f; c2 = 0.f; }
      float eh0 = 0.f, eh1 = 0.f, eh2 = 0.f;
      if (active) {
        eh0 = thr0 * em_hit[gi * 3 + 0];
        eh1 = thr1 * em_hit[gi * 3 + 1];
        eh2 = thr2 * em_hit[gi * 3 + 2];
      }
      res0 += eh0; res1 += eh1; res2 += eh2;
      const bool active_next = ((depth + 1) < DD) && valid;
      if (active && !active_next) { res0 += thr0 * c0; res1 += thr1 * c1; res2 += thr2 * c2; }
      if (d == 0 && active && active_next) { L0 = c0 + eh0; L1 = c1 + eh1; L2 = c2 + eh2; }
      if (active_next) {
        res0 += thr0 * em_sample[gi * 3 + 0];
        res1 += thr1 * em_sample[gi * 3 + 1];
        res2 += thr2 * em_sample[gi * 3 + 2];
        thr0 *= bsdf_weight[gi * 3 + 0];
        thr1 *= bsdf_weight[gi * 3 + 1];
        thr2 *= bsdf_weight[gi * 3 + 2];
        eta  *= eta_s[gi];
      }
      validray = validray || (active && valid && !nullf);
      depth += valid ? 1 : 0;
      const float tmax    = fmaxf(thr0, fmaxf(thr1, thr2));
      const float rr_prob = fminf(tmax * eta * eta, 0.95f);
      const bool  rr_act  = depth >= RRD;
      const bool  rr_cont = rr_u[gi] < rr_prob;
      const float inv     = (rr_prob > 0.f) ? (1.f / fmaxf(rr_prob, 1e-20f)) : 0.f;
      if (rr_act) { thr0 *= inv; thr1 *= inv; thr2 *= inv; }
      active = active_next && (!rr_act || rr_cont) && (tmax != 0.f);
    }
  }

  // ---- outputs: rgb | aov | residual, each N x 3, concatenated flat ----
  if (owner && ray < N) {
    const long long o  = (long long)ray * 3;
    const long long NN = (long long)N * 3;
    out[o + 0] = validray ? res0 : 0.f;
    out[o + 1] = validray ? res1 : 0.f;
    out[o + 2] = validray ? res2 : 0.f;
    out[NN + o + 0] = validray ? L0 : 0.f;
    out[NN + o + 1] = validray ? L1 : 0.f;
    out[NN + o + 2] = validray ? L2 : 0.f;
    const float d0 = L0 - res0, d1 = L1 - res1, d2 = L2 - res2;
    out[2 * NN + o + 0] = d0 * d0;
    out[2 * NN + o + 1] = d1 * d1;
    out[2 * NN + o + 2] = d2 * d2;
  }
}

extern "C" void kernel_launch(void* const* d_in, const int* in_sizes, int n_in,
                              void* d_out, int out_size, void* d_ws, size_t ws_size,
                              hipStream_t stream) {
  (void)n_in; (void)out_size; (void)ws_size;
  const float* si_p        = (const float*)d_in[0];
  const float* si_wo       = (const float*)d_in[1];
  const float* si_n        = (const float*)d_in[2];
  const float* albedo      = (const float*)d_in[3];
  const float* em_hit      = (const float*)d_in[4];
  const float* em_sample   = (const float*)d_in[5];
  const float* bsdf_weight = (const float*)d_in[6];
  const float* eta_s       = (const float*)d_in[7];
  const float* rr_u        = (const float*)d_in[8];
  const float* W0 = (const float*)d_in[9];
  const float* b0 = (const float*)d_in[10];
  const float* W1 = (const float*)d_in[11];
  const float* b1 = (const float*)d_in[12];
  const float* W2 = (const float*)d_in[13];
  const float* b2 = (const float*)d_in[14];
  const float* W3 = (const float*)d_in[15];
  const float* b3 = (const float*)d_in[16];
  const unsigned char* si_valid  = (const unsigned char*)d_in[17];
  const unsigned char* null_flag = (const unsigned char*)d_in[18];

  const int N = in_sizes[7] / DD;   // eta_s has D*N elements

  // workspace: transposed f16 weights (A-operand layout, row = out-neuron)
  _Float16* W0t = (_Float16*)d_ws;          // 256*32
  _Float16* W1t = W0t + 256 * 32;           // 256*256
  _Float16* W2t = W1t + 256 * 256;          // 256*256
  _Float16* W3t = W2t + 256 * 256;          // 16*256

  prep_w0_k<<<(256 * 32  + 255) / 256, 256, 0, stream>>>(W0, W0t);
  prep_w_k <<<(256 * 256 + 255) / 256, 256, 0, stream>>>(W1, W1t);
  prep_w_k <<<(256 * 256 + 255) / 256, 256, 0, stream>>>(W2, W2t);
  prep_w3_k<<<(16 * 256  + 255) / 256, 256, 0, stream>>>(W3, W3t);

  const int raysPerBlock = WPB * RPW;       // 128
  const int blocks = (N + raysPerBlock - 1) / raysPerBlock;
  nerad_reg_k<<<blocks, THREADS, 0, stream>>>(
      si_p, si_wo, si_n, albedo, em_hit, em_sample, bsdf_weight, eta_s, rr_u,
      b0, b1, b2, b3, W0t, W1t, W2t, W3t, si_valid, null_flag,
      (float*)d_out, N);
}
